// DGCNN_cls_58488864637077
// MI455X (gfx1250) — matrix operand order repaired and checked by reading, not compiled, not run on hardware
//
#include <hip/hip_runtime.h>
#include <hip/hip_bf16.h>

typedef __attribute__((ext_vector_type(16))) _Float16 v16h;
typedef __attribute__((ext_vector_type(8)))  float    v8f;

#define BN_EPS 1e-5f

// =====================================================================
// WMMA GEMM:  C = A * B^T   (f32 in memory, f16 WMMA compute)
//   A : [M, K]     row-major  (+ batch stride sAb)
//   Bm: [Nout, K]  row-major  (+ batch stride sBb)
//   C : [M, Nout]  row-major  (+ batch stride sCb)
// REQUIRES: K % 32 == 0, M % 64 == 0, Nout % 64 == 0 (callers guarantee).
// Tiles staged f32 into LDS with gfx1250 async DMA
// (global_load_async_to_lds_b128, ASYNCcnt), converted to f16 during the
// per-lane fragment gather. Block 256 thr = 8 waves; 64x64 C tile; each
// wave owns two 16x16 tiles sharing the A fragment (2 wmma / K-step).
// Fused inference-BN (bn=[gamma;beta]) and ReLU epilogue.
// =====================================================================
__global__ __launch_bounds__(256)
void gemm_nt_wmma(const float* __restrict__ A, const float* __restrict__ Bm,
                  float* __restrict__ C, int M, int Nout, int K,
                  long long sAb, long long sBb, long long sCb,
                  const float* __restrict__ bn, int relu)
{
    __shared__ __align__(16) float Asf[64][36];   // 64 rows x 32 K, padded stride
    __shared__ __align__(16) float Bsf[64][36];   // 64 cols x 32 K, padded stride

    const int bm  = blockIdx.x;      // M tile (64 rows)
    const int bnn = blockIdx.y;      // N tile (64 cols)
    const int bz  = blockIdx.z;      // batch
    const float* Ab = A  + (long long)bz * sAb + (long long)(bm  * 64) * K;
    const float* Bb = Bm + (long long)bz * sBb + (long long)(bnn * 64) * K;
    float*       Cb = C  + (long long)bz * sCb;

    const int tid  = threadIdx.x;
    const int lane = tid & 31;
    const int wave = tid >> 5;       // 0..7
    const int wr   = wave & 3;       // 16-row strip of 64
    const int wc   = wave >> 2;      // 32-col strip of 64
    const int lg   = lane >> 4;      // lane group 0/1
    const int mr   = lane & 15;

    // staging coords: thread copies one 16B chunk per 32-row half tile
    const int sr = tid >> 3;         // row 0..31
    const int sc = (tid & 7) * 4;    // f32 col 0,4,...,28

    const unsigned asBase = (unsigned)(size_t)(&Asf[0][0]);
    const unsigned bsBase = (unsigned)(size_t)(&Bsf[0][0]);
    const unsigned aOff0 = asBase + (unsigned)((sr      ) * 36 + sc) * 4u;
    const unsigned aOff1 = asBase + (unsigned)((sr + 32) * 36 + sc) * 4u;
    const unsigned bOff0 = bsBase + (unsigned)((sr      ) * 36 + sc) * 4u;
    const unsigned bOff1 = bsBase + (unsigned)((sr + 32) * 36 + sc) * 4u;

    v8f acc0 = {}, acc1 = {};

    for (int k0 = 0; k0 < K; k0 += 32) {
        const float* ga0 = Ab + (long long)(sr     ) * K + k0 + sc;
        const float* ga1 = Ab + (long long)(sr + 32) * K + k0 + sc;
        const float* gb0 = Bb + (long long)(sr     ) * K + k0 + sc;
        const float* gb1 = Bb + (long long)(sr + 32) * K + k0 + sc;
        // gfx1250 async DMA: global -> LDS, tracked by ASYNCcnt
        asm volatile("global_load_async_to_lds_b128 %0, %1, off"
                     :: "v"(aOff0), "v"(ga0) : "memory");
        asm volatile("global_load_async_to_lds_b128 %0, %1, off"
                     :: "v"(aOff1), "v"(ga1) : "memory");
        asm volatile("global_load_async_to_lds_b128 %0, %1, off"
                     :: "v"(bOff0), "v"(gb0) : "memory");
        asm volatile("global_load_async_to_lds_b128 %0, %1, off"
                     :: "v"(bOff1), "v"(gb1) : "memory");
        asm volatile("s_wait_asynccnt 0x0" ::: "memory");
        __syncthreads();

        // per-lane fragment gather per CDNA5 16-bit WMMA VGPR layouts
        v16h af, bf0, bf1;
        #pragma unroll
        for (int h = 0; h < 16; ++h) {
            int v  = h >> 1;
            // A 16x32: lanes 0-15 cover K {0..7,16..23}; lanes 16-31 {8..15,24..31}
            int ka = h + (v < 4 ? (lg ? 8 : 0) : (lg ? 16 : 8));
            af[h]  = (_Float16)Asf[wr * 16 + mr][ka];
            // B 32x16: same K order as A (lanes 0-15 {0..7,16..23}; lanes 16-31 {8..15,24..31})
            int kb = h + (h < 8 ? lg * 8 : 8 + lg * 8);
            bf0[h] = (_Float16)Bsf[wc * 32 + mr][kb];
            bf1[h] = (_Float16)Bsf[wc * 32 + 16 + mr][kb];
        }
        acc0 = __builtin_amdgcn_wmma_f32_16x16x32_f16(false, af, false, bf0,
                                                      (short)0, acc0, false, false);
        acc1 = __builtin_amdgcn_wmma_f32_16x16x32_f16(false, af, false, bf1,
                                                      (short)0, acc1, false, false);
        __syncthreads();
    }

    // epilogue: C 16x16 f32 layout (VGPR r: lanes0-15 -> M=r, lanes16-31 -> M=r+8)
    const float rsq = rsqrtf(1.0f + BN_EPS);
    #pragma unroll
    for (int r = 0; r < 8; ++r) {
        int row  = bm * 64 + wr * 16 + lg * 8 + r;
        int col0 = bnn * 64 + wc * 32 + mr;
        int col1 = col0 + 16;
        if (row < M) {
            float y0 = acc0[r], y1 = acc1[r];
            if (bn) {
                y0 = y0 * (bn[col0] * rsq) + bn[Nout + col0];
                y1 = y1 * (bn[col1] * rsq) + bn[Nout + col1];
            }
            if (relu) { y0 = fmaxf(y0, 0.0f); y1 = fmaxf(y1, 0.0f); }
            Cb[(long long)row * Nout + col0] = y0;
            Cb[(long long)row * Nout + col1] = y1;
        }
    }
}

// =====================================================================
// Elementwise / reduction helper kernels
// =====================================================================
__global__ void k_transpose_x(const float* __restrict__ x, float* __restrict__ f,
                              int C, int Nn, long long total)
{   // x [B,C,N] -> f [B,N,C]
    long long i = (long long)blockIdx.x * blockDim.x + threadIdx.x;
    if (i >= total) return;
    int c = (int)(i % C);
    long long bn = i / C;
    int n = (int)(bn % Nn);
    int b = (int)(bn / Nn);
    f[i] = x[((long long)b * C + c) * Nn + n];
}

__global__ void k_transpose_w(const float* __restrict__ w, float* __restrict__ wt,
                              int Kin, int Out, int Kpad)
{   // w [Kin,Out] -> wt [Out,Kpad], zero-padded K tail
    long long i = (long long)blockIdx.x * blockDim.x + threadIdx.x;
    if (i >= (long long)Out * Kpad) return;
    int k = (int)(i % Kpad);
    int o = (int)(i / Kpad);
    wt[i] = (k < Kin) ? w[(long long)k * Out + o] : 0.0f;
}

__global__ void k_d2_c6(const float* __restrict__ f, float* __restrict__ d2,
                        int Nn, long long total)
{   // d2[b,n,m] = |fn|^2 + |fm|^2 - 2 fn.fm   for C=6
    long long i = (long long)blockIdx.x * blockDim.x + threadIdx.x;
    if (i >= total) return;
    int m = (int)(i % Nn);
    long long r = i / Nn;                 // b*Nn + n
    int b = (int)(r / Nn);
    const float* fn = f + r * 6;
    const float* fm = f + ((long long)b * Nn + m) * 6;
    float sn = 0.f, sm = 0.f, dt = 0.f;
    #pragma unroll
    for (int c = 0; c < 6; ++c) {
        sn += fn[c] * fn[c];
        sm += fm[c] * fm[c];
        dt += fn[c] * fm[c];
    }
    d2[i] = sn + sm - 2.0f * dt;
}

__global__ void k_rr(const float* __restrict__ f, float* __restrict__ rr,
                     int C, int total)
{
    int i = blockIdx.x * blockDim.x + threadIdx.x;
    if (i >= total) return;
    const float* p = f + (long long)i * C;
    float s = 0.f;
    for (int c = 0; c < C; ++c) s += p[c] * p[c];
    rr[i] = s;
}

__global__ void k_d2g(const float* __restrict__ G, const float* __restrict__ rr,
                      float* __restrict__ out, int Nn, int clamp, long long total)
{   // out[b,n,m] = rr[b,n] + rr[b,m] - 2 G[b,n,m]  (optional clamp). In-place safe.
    long long i = (long long)blockIdx.x * blockDim.x + threadIdx.x;
    if (i >= total) return;
    int m = (int)(i % Nn);
    long long r = i / Nn;
    int b = (int)(r / Nn);
    float v = rr[r] + rr[(long long)b * Nn + m] - 2.0f * G[i];
    if (clamp) v = fmaxf(v, 0.0f);
    out[i] = v;
}

__global__ void k_topk20(const float* __restrict__ d2, int* __restrict__ idx,
                         int Nn, int totalRows)
{   // 20 smallest per row, index-ascending tiebreak (matches top_k of -d2)
    int row = blockIdx.x * blockDim.x + threadIdx.x;
    if (row >= totalRows) return;
    const float* d = d2 + (long long)row * Nn;
    float bd[20]; int bi[20];
    #pragma unroll
    for (int j = 0; j < 20; ++j) { bd[j] = 3.4e38f; bi[j] = 0; }
    for (int m = 0; m < Nn; ++m) {
        float v = d[m];
        if (v < bd[19]) {
            int j = 19;
            while (j > 0 && bd[j - 1] > v) {
                bd[j] = bd[j - 1]; bi[j] = bi[j - 1]; --j;
            }
            bd[j] = v; bi[j] = m;
        }
    }
    #pragma unroll
    for (int j = 0; j < 20; ++j) idx[(long long)row * 20 + j] = bi[j];
}

__global__ void k_edge(const float* __restrict__ f, const int* __restrict__ idx,
                       float* __restrict__ e, int C, int Cpad, int Nn,
                       long long total)
{   // e[row, 0:C]=f[nb]-f[n]; e[row, C:2C]=f[n]; e[row, 2C:Cpad]=0
    long long i = (long long)blockIdx.x * blockDim.x + threadIdx.x;
    if (i >= total) return;
    int c2 = (int)(i % Cpad);
    long long t = i / Cpad;
    int k = (int)(t % 20);
    long long bn = t / 20;               // b*Nn + n
    int b = (int)(bn / Nn);
    float v;
    if (c2 < C) {
        int nb = idx[bn * 20 + k];
        v = f[((long long)b * Nn + nb) * C + c2] - f[bn * C + c2];
    } else if (c2 < 2 * C) {
        v = f[bn * C + (c2 - C)];
    } else {
        v = 0.0f;
    }
    e[i] = v;
}

__global__ void k_maxk(const float* __restrict__ h, float* __restrict__ out,
                       int C, long long total)
{   // h [rows,20,C] -> out [rows,C] max over k
    long long i = (long long)blockIdx.x * blockDim.x + threadIdx.x;
    if (i >= total) return;
    int c = (int)(i % C);
    long long bn = i / C;
    const float* p = h + bn * 20 * C + c;
    float m = p[0];
    #pragma unroll
    for (int k = 1; k < 20; ++k) m = fmaxf(m, p[(long long)k * C]);
    out[i] = m;
}

__global__ void k_concat3(const float* __restrict__ x1, const float* __restrict__ x2,
                          const float* __restrict__ x3, float* __restrict__ cat,
                          long long total)
{
    long long i = (long long)blockIdx.x * blockDim.x + threadIdx.x;
    if (i >= total) return;
    int c = (int)(i % 192);
    long long bn = i / 192;
    float v;
    if      (c < 64)  v = x1[bn * 64 + c];
    else if (c < 128) v = x2[bn * 64 + (c - 64)];
    else              v = x3[bn * 64 + (c - 128)];
    cat[i] = v;
}

__global__ void k_gmax(const float* __restrict__ h5, float* __restrict__ g,
                       int Nn, int total)
{   // h5 [B,N,1024] -> g [B,1024] max over N
    int i = blockIdx.x * blockDim.x + threadIdx.x;
    if (i >= total) return;
    int c = i % 1024;
    int b = i / 1024;
    float m = -3.4e38f;
    for (int n = 0; n < Nn; ++n)
        m = fmaxf(m, h5[((long long)b * Nn + n) * 1024 + c]);
    g[i] = m;
}

__global__ void k_concat_expand(const float* __restrict__ g, const float* __restrict__ x1,
                                const float* __restrict__ x2, const float* __restrict__ x3,
                                float* __restrict__ net, int Nn, long long total)
{
    long long i = (long long)blockIdx.x * blockDim.x + threadIdx.x;
    if (i >= total) return;
    int c = (int)(i % 1216);
    long long bn = i / 1216;
    int b = (int)(bn / Nn);
    float v;
    if      (c < 1024) v = g[(long long)b * 1024 + c];
    else if (c < 1088) v = x1[bn * 64 + (c - 1024)];
    else if (c < 1152) v = x2[bn * 64 + (c - 1088)];
    else               v = x3[bn * 64 + (c - 1152)];
    net[i] = v;
}

__global__ void k_center(const float* __restrict__ c128, const float* __restrict__ w9,
                         float* __restrict__ cs, int total)
{
    int i = blockIdx.x * blockDim.x + threadIdx.x;
    if (i >= total) return;
    const float* r = c128 + (long long)i * 128;
    float s = 0.f;
    #pragma unroll 8
    for (int c = 0; c < 128; ++c) s += r[c] * w9[c];
    cs[i] = 1.0f / (1.0f + expf(-s));
}

// =====================================================================
// Host orchestration
// =====================================================================
static inline int nblk(long long total, int bs) { return (int)((total + bs - 1) / bs); }

extern "C" void kernel_launch(void* const* d_in, const int* in_sizes, int n_in,
                              void* d_out, int out_size, void* d_ws, size_t ws_size,
                              hipStream_t stream)
{
    (void)in_sizes; (void)n_in; (void)out_size; (void)ws_size;
    constexpr int B = 4, Nn = 2048, Kn = 20;
    constexpr long long Mrows = (long long)B * Nn;          // 8192
    constexpr long long Medge = Mrows * Kn;                 // 163840

    const float* x    = (const float*)d_in[0];
    const float* w1   = (const float*)d_in[1];  const float* bn1   = (const float*)d_in[2];
    const float* w1_1 = (const float*)d_in[3];  const float* bn1_1 = (const float*)d_in[4];
    const float* w2   = (const float*)d_in[5];  const float* bn2   = (const float*)d_in[6];
    const float* w2_2 = (const float*)d_in[7];  const float* bn2_2 = (const float*)d_in[8];
    const float* w3   = (const float*)d_in[9];  const float* bn3   = (const float*)d_in[10];
    const float* w5   = (const float*)d_in[11]; const float* bn5   = (const float*)d_in[12];
    const float* w6   = (const float*)d_in[13]; const float* bn6   = (const float*)d_in[14];
    const float* w7   = (const float*)d_in[15]; const float* bn7   = (const float*)d_in[16];
    const float* w8   = (const float*)d_in[17];
    const float* w9   = (const float*)d_in[18];
    const float* w11  = (const float*)d_in[19];

    // ---- output layout: center_score | fsim | simmat | p_distance ----
    float* out    = (float*)d_out;
    float* cs     = out;
    float* fsim   = out + Mrows;
    float* simmat = fsim + Mrows * 128;
    float* pdist  = simmat + Mrows * Nn;

    // ---- workspace arena ----
    float* ws = (float*)d_ws;
    size_t o = 0;
    auto alloc = [&](size_t n) { float* p = ws + o; o += (n + 3) & ~(size_t)3; return p; };
    float* f0    = alloc((size_t)Mrows * 6);
    float* x1    = alloc((size_t)Mrows * 64);
    float* x2    = alloc((size_t)Mrows * 64);
    float* x3    = alloc((size_t)Mrows * 64);
    float* rrb   = alloc((size_t)Mrows);
    int*   idx   = (int*)alloc((size_t)Mrows * Kn);
    float* cat   = alloc((size_t)Mrows * 192);
    float* gbuf  = alloc((size_t)B * 1024);
    float* w1T   = alloc(64 * 32);           // padded K: 12 -> 32
    float* w1_1T = alloc(64 * 64);
    float* w2T   = alloc(64 * 128);
    float* w2_2T = alloc(64 * 64);
    float* w3T   = alloc(64 * 128);
    float* w5T   = alloc(1024 * 192);
    float* w6T   = alloc(512 * 1216);
    float* w7T   = alloc(256 * 512);
    float* w8T   = alloc(128 * 256);
    float* w11T  = alloc(128 * 256);
    float* c128  = alloc((size_t)Mrows * 128);
    float* h5    = alloc((size_t)Mrows * 1024);
    float* net1216 = alloc((size_t)Mrows * 1216);
    float* net512  = alloc((size_t)Mrows * 512);
    float* net256  = alloc((size_t)Mrows * 256);
    float* hA    = alloc((size_t)Medge * 64);
    float* hB    = alloc((size_t)Medge * 64);
    float* arena = alloc((size_t)Medge * 128);   // Gram / d2 / edge-feature union

    const int BS = 256;
    auto gemmGrid = [](long long M, int Nout, int batch) {
        return dim3((unsigned)((M + 63) / 64), (unsigned)((Nout + 63) / 64), (unsigned)batch);
    };

    // ---- preprocessing: transpose input + all weights (K zero-padded) ----
    k_transpose_x<<<nblk(Mrows * 6, BS), BS, 0, stream>>>(x, f0, 6, Nn, Mrows * 6);
    k_transpose_w<<<nblk(64 * 32, BS),    BS, 0, stream>>>(w1,   w1T,   12,   64,   32);
    k_transpose_w<<<nblk(64 * 64, BS),    BS, 0, stream>>>(w1_1, w1_1T, 64,   64,   64);
    k_transpose_w<<<nblk(64 * 128, BS),   BS, 0, stream>>>(w2,   w2T,   128,  64,   128);
    k_transpose_w<<<nblk(64 * 64, BS),    BS, 0, stream>>>(w2_2, w2_2T, 64,   64,   64);
    k_transpose_w<<<nblk(64 * 128, BS),   BS, 0, stream>>>(w3,   w3T,   128,  64,   128);
    k_transpose_w<<<nblk(1024 * 192, BS), BS, 0, stream>>>(w5,   w5T,   192,  1024, 192);
    k_transpose_w<<<nblk(512 * 1216, BS), BS, 0, stream>>>(w6,   w6T,   1216, 512,  1216);
    k_transpose_w<<<nblk(256 * 512, BS),  BS, 0, stream>>>(w7,   w7T,   512,  256,  512);
    k_transpose_w<<<nblk(128 * 256, BS),  BS, 0, stream>>>(w8,   w8T,   256,  128,  256);
    k_transpose_w<<<nblk(128 * 256, BS),  BS, 0, stream>>>(w11,  w11T,  256,  128,  256);

    const long long NNtot = Mrows * Nn;

    // ================= EdgeConv block 1 (C=6, padded to K=32) =================
    k_d2_c6<<<nblk(NNtot, BS), BS, 0, stream>>>(f0, pdist, Nn, NNtot);  // p_distance
    k_topk20<<<nblk(Mrows, BS), BS, 0, stream>>>(pdist, idx, Nn, (int)Mrows);
    k_edge<<<nblk(Medge * 32, BS), BS, 0, stream>>>(f0, idx, arena, 6, 32, Nn, Medge * 32);
    gemm_nt_wmma<<<gemmGrid(Medge, 64, 1), BS, 0, stream>>>(arena, w1T, hA,
        (int)Medge, 64, 32, 0, 0, 0, bn1, 1);
    gemm_nt_wmma<<<gemmGrid(Medge, 64, 1), BS, 0, stream>>>(hA, w1_1T, hB,
        (int)Medge, 64, 64, 0, 0, 0, bn1_1, 1);
    k_maxk<<<nblk(Mrows * 64, BS), BS, 0, stream>>>(hB, x1, 64, Mrows * 64);

    // ================= EdgeConv block 2 (C=64) =================
    gemm_nt_wmma<<<gemmGrid(Nn, Nn, B), BS, 0, stream>>>(x1, x1, arena,
        Nn, Nn, 64, (long long)Nn * 64, (long long)Nn * 64, (long long)Nn * Nn, nullptr, 0);
    k_rr<<<nblk(Mrows, BS), BS, 0, stream>>>(x1, rrb, 64, (int)Mrows);
    k_d2g<<<nblk(NNtot, BS), BS, 0, stream>>>(arena, rrb, arena, Nn, 0, NNtot);
    k_topk20<<<nblk(Mrows, BS), BS, 0, stream>>>(arena, idx, Nn, (int)Mrows);
    k_edge<<<nblk(Medge * 128, BS), BS, 0, stream>>>(x1, idx, arena, 64, 128, Nn, Medge * 128);
    gemm_nt_wmma<<<gemmGrid(Medge, 64, 1), BS, 0, stream>>>(arena, w2T, hA,
        (int)Medge, 64, 128, 0, 0, 0, bn2, 1);
    gemm_nt_wmma<<<gemmGrid(Medge, 64, 1), BS, 0, stream>>>(hA, w2_2T, hB,
        (int)Medge, 64, 64, 0, 0, 0, bn2_2, 1);
    k_maxk<<<nblk(Mrows * 64, BS), BS, 0, stream>>>(hB, x2, 64, Mrows * 64);

    // ================= EdgeConv block 3 (C=64) =================
    gemm_nt_wmma<<<gemmGrid(Nn, Nn, B), BS, 0, stream>>>(x2, x2, arena,
        Nn, Nn, 64, (long long)Nn * 64, (long long)Nn * 64, (long long)Nn * Nn, nullptr, 0);
    k_rr<<<nblk(Mrows, BS), BS, 0, stream>>>(x2, rrb, 64, (int)Mrows);
    k_d2g<<<nblk(NNtot, BS), BS, 0, stream>>>(arena, rrb, arena, Nn, 0, NNtot);
    k_topk20<<<nblk(Mrows, BS), BS, 0, stream>>>(arena, idx, Nn, (int)Mrows);
    k_edge<<<nblk(Medge * 128, BS), BS, 0, stream>>>(x2, idx, arena, 64, 128, Nn, Medge * 128);
    gemm_nt_wmma<<<gemmGrid(Medge, 64, 1), BS, 0, stream>>>(arena, w3T, hA,
        (int)Medge, 64, 128, 0, 0, 0, bn3, 1);
    k_maxk<<<nblk(Mrows * 64, BS), BS, 0, stream>>>(hA, x3, 64, Mrows * 64);

    // ================= Global feature + point MLPs =================
    k_concat3<<<nblk(Mrows * 192, BS), BS, 0, stream>>>(x1, x2, x3, cat, Mrows * 192);
    gemm_nt_wmma<<<gemmGrid(Mrows, 1024, 1), BS, 0, stream>>>(cat, w5T, h5,
        (int)Mrows, 1024, 192, 0, 0, 0, bn5, 1);
    k_gmax<<<nblk(B * 1024, BS), BS, 0, stream>>>(h5, gbuf, Nn, B * 1024);
    k_concat_expand<<<nblk(Mrows * 1216, BS), BS, 0, stream>>>(gbuf, x1, x2, x3,
        net1216, Nn, Mrows * 1216);
    gemm_nt_wmma<<<gemmGrid(Mrows, 512, 1), BS, 0, stream>>>(net1216, w6T, net512,
        (int)Mrows, 512, 1216, 0, 0, 0, bn6, 1);
    gemm_nt_wmma<<<gemmGrid(Mrows, 256, 1), BS, 0, stream>>>(net512, w7T, net256,
        (int)Mrows, 256, 512, 0, 0, 0, bn7, 1);

    // ================= Heads =================
    gemm_nt_wmma<<<gemmGrid(Mrows, 128, 1), BS, 0, stream>>>(net256, w8T, c128,
        (int)Mrows, 128, 256, 0, 0, 0, nullptr, 1);
    k_center<<<nblk(Mrows, BS), BS, 0, stream>>>(c128, w9, cs, (int)Mrows);

    gemm_nt_wmma<<<gemmGrid(Mrows, 128, 1), BS, 0, stream>>>(net256, w11T, fsim,
        (int)Mrows, 128, 256, 0, 0, 0, nullptr, 1);
    k_rr<<<nblk(Mrows, BS), BS, 0, stream>>>(fsim, rrb, 128, (int)Mrows);
    gemm_nt_wmma<<<gemmGrid(Nn, Nn, B), BS, 0, stream>>>(fsim, fsim, arena,
        Nn, Nn, 128, (long long)Nn * 128, (long long)Nn * 128, (long long)Nn * Nn, nullptr, 0);
    k_d2g<<<nblk(NNtot, BS), BS, 0, stream>>>(arena, rrb, simmat, Nn, 1, NNtot);
}
